// Performer_Attention_18098992185935
// MI455X (gfx1250) — matrix operand order repaired and checked by reading, not compiled, not run on hardware
//
#include <hip/hip_runtime.h>
#include <cstddef>

// ---------------------------------------------------------------------------
// Performer attention on MI455X (gfx1250, wave32, WMMA).
// Linear-attention rewrite: attn rows are positive (exp features), so the
// L1 denominator == qf . (sum_n kf). All GEMMs use v_wmma_f32_16x16x32_f16.
// GEMMs are "NT": A row-major [M,K], B stored row-major as [Ncols,K], so every
// lane's fragment is a contiguous 16B chunk (global_load_b128 / ds_load_b128).
// Big GEMMs (qkv, proj): 4-wave workgroups stage the shared 16xK A-tile into
// LDS with global_load_async_to_lds_b128 (ASYNCcnt path), B fragments are
// register double-buffered so loads overlap WMMA issue.
// ---------------------------------------------------------------------------

#define BB 8
#define NN 1024
#define CC 768
#define HH 12
#define DD 64
#define MM 128

#define SQ   0.3535533905932738f   /* D^-0.25 */
#define MINV 0.08838834764831845f  /* M^-0.5  */

typedef __attribute__((ext_vector_type(16))) _Float16 v16h;
typedef __attribute__((ext_vector_type(8)))  float    v8f;

union Frag { v16h v; float4 f[2]; };

#if defined(__has_builtin)
#if __has_builtin(__builtin_amdgcn_global_load_async_to_lds_b128)
#define HAVE_ASYNC_LDS 1
#endif
#endif

#ifdef HAVE_ASYNC_LDS
// Match the builtin's parameter types exactly (per hipcc diagnostic):
//   arg0: vector_size(16) int __device__(AS1) pointer (global source)
//   arg1: vector_size(16) int __shared__(AS3) pointer (LDS destination)
typedef int vi4 __attribute__((vector_size(16)));
typedef vi4 __attribute__((address_space(1)))* gvi4p;
typedef vi4 __attribute__((address_space(3)))* lvi4p;
#endif

// A fragment: 16x32 f16 tile. Lane layout (ISA 7.12.2, 16-bit A 16x32):
//   row = lane&15 ; v0..v3 hold K = k0+(lane>>4)*8 .. +7 ; v4..v7 hold +16.
// Works for global or LDS-backed pointers (compiler picks global/ds b128 ops).
__device__ __forceinline__ v16h load_a(const _Float16* __restrict__ A, int lda,
                                       int k0, int lane) {
    const _Float16* p = A + (size_t)(lane & 15) * lda + k0 + ((lane >> 4) << 3);
    Frag fr;
    fr.f[0] = *(const float4*)p;          // K offsets 0..7
    fr.f[1] = *(const float4*)(p + 16);   // K offsets 16..23
    return fr.v;
}

// B fragment: 32x16 f16 tile from Bt stored [Ncols, K] row-major.
//   col = lane&15 ; v0..v3 hold K = k0+(lane>>4)*8 .. +7 ; v4..v7 hold +16.
__device__ __forceinline__ v16h load_b(const _Float16* __restrict__ Bt, int ldb,
                                       int k0, int lane) {
    const _Float16* p = Bt + (size_t)(lane & 15) * ldb + k0 + ((lane >> 4) << 3);
    Frag fr;
    fr.f[0] = *(const float4*)p;
    fr.f[1] = *(const float4*)(p + 16);
    return fr.v;
}

__device__ __forceinline__ v8f wmma32(v16h a, v16h b, v8f c) {
    return __builtin_amdgcn_wmma_f32_16x16x32_f16(false, a, false, b,
                                                  (short)0, c, false, false);
}

// Stage `nhalfs` f16 (multiple of 8*blockDim) from global into LDS.
__device__ __forceinline__ void stage_tile(const _Float16* __restrict__ gsrc,
                                           _Float16* lds_dst, int nhalfs,
                                           int tid, int nthreads) {
#ifdef HAVE_ASYNC_LDS
    for (int i = tid * 8; i < nhalfs; i += nthreads * 8)
        __builtin_amdgcn_global_load_async_to_lds_b128(
            (gvi4p)(gsrc + i), (lvi4p)(lds_dst + i), 0, 0);
#if __has_builtin(__builtin_amdgcn_s_wait_asynccnt)
    __builtin_amdgcn_s_wait_asynccnt(0);
#else
    asm volatile("s_wait_asynccnt 0x0" ::: "memory");
#endif
#else
    for (int i = tid * 8; i < nhalfs; i += nthreads * 8)
        *(float4*)(lds_dst + i) = *(const float4*)(gsrc + i);
#endif
    __syncthreads();
}

// ---------------------------------------------------------------------------
// Elementwise converters
// ---------------------------------------------------------------------------
__global__ void k_cvt(const float* __restrict__ src, _Float16* __restrict__ dst, int n) {
    int i = blockIdx.x * 256 + threadIdx.x;
    if (i < n) dst[i] = (_Float16)src[i];
}

// rand_matrix [H][D][M] -> Rth [H][M][D] (f16), so feature GEMMs are K-contiguous
__global__ void k_cvt_randT(const float* __restrict__ R, _Float16* __restrict__ Rt) {
    int i = blockIdx.x * 256 + threadIdx.x;
    if (i >= HH * DD * MM) return;
    int h = i / (DD * MM);
    int rem = i % (DD * MM);
    int d = rem / MM, m = rem % MM;
    Rt[((size_t)h * MM + m) * DD + d] = (_Float16)R[i];
}

// ---------------------------------------------------------------------------
// qkv GEMM: [8192,768] x [2304,768]^T. 4 waves/block; block tile 16x256
// (wave tile 16x64). A tile (16x768, 24KB) staged in LDS; B double-buffered.
// Scatters q,k (pre-scaled by D^-1/4) into [b,h,n,d] and v transposed [b,h,d,n].
// ---------------------------------------------------------------------------
__global__ void k_qkv_gemm(const _Float16* __restrict__ Xh, const _Float16* __restrict__ Wh,
                           _Float16* __restrict__ qh, _Float16* __restrict__ kh,
                           _Float16* __restrict__ vT) {
    __shared__ __align__(16) _Float16 As[16 * CC];
    int lane = threadIdx.x & 31, wid = threadIdx.x >> 5;
    int row0 = blockIdx.x * 16;               // 512 row tiles
    int col0 = blockIdx.y * 256 + wid * 64;   // 9 col superblocks x 4 waves
    stage_tile(Xh + (size_t)row0 * CC, As, 16 * CC, threadIdx.x, 128);

    const _Float16* Bt[4];
#pragma unroll
    for (int j = 0; j < 4; ++j) Bt[j] = Wh + (size_t)(col0 + j * 16) * CC;

    v16h b[4], nb[4];
#pragma unroll
    for (int j = 0; j < 4; ++j) b[j] = load_b(Bt[j], CC, 0, lane);
    v8f acc[4] = {};
    for (int k0 = 0; k0 < CC; k0 += 32) {
        int kn = (k0 + 32 < CC) ? k0 + 32 : 0;
#pragma unroll
        for (int j = 0; j < 4; ++j) nb[j] = load_b(Bt[j], CC, kn, lane);
        v16h a = load_a(As, CC, k0, lane);    // ds_load_b128 from LDS
#pragma unroll
        for (int j = 0; j < 4; ++j) {
            acc[j] = wmma32(a, b[j], acc[j]);
            b[j] = nb[j];
        }
    }
    int hl = lane >> 4, c = lane & 15;
#pragma unroll
    for (int j = 0; j < 4; ++j) {
        int o = col0 + j * 16 + c;
        int three = o / CC, rem = o % CC;
        int h = rem / DD, d = rem % DD;
#pragma unroll
        for (int i = 0; i < 8; ++i) {
            int r = row0 + i + (hl << 3);
            int bb = r >> 10, n = r & (NN - 1);
            float val = acc[j][i];
            size_t bh = (size_t)bb * HH + h;
            if (three == 0)      qh[(bh * NN + n) * DD + d] = (_Float16)(val * SQ);
            else if (three == 1) kh[(bh * NN + n) * DD + d] = (_Float16)(val * SQ);
            else                 vT[(bh * DD + d) * NN + n] = (_Float16)val;
        }
    }
}

// ---------------------------------------------------------------------------
// qf = exp(z @ R - 0.5||z||^2) * M^-0.5   stored [b,h,n,m] (K=m contiguous)
// ---------------------------------------------------------------------------
__global__ void k_qf(const _Float16* __restrict__ qh, const _Float16* __restrict__ Rth,
                     _Float16* __restrict__ qf) {
    int lane = threadIdx.x;
    int id = blockIdx.x;               // B*H*(N/16) = 6144
    int ntile = id & 63; id >>= 6;
    int h = id % HH, b = id / HH;
    const _Float16* A  = qh + (((size_t)b * HH + h) * NN + ntile * 16) * DD;
    const _Float16* Rt = Rth + (size_t)h * MM * DD;
    __shared__ float sqn[16];
    if (lane < 16) {
        const _Float16* zr = A + (size_t)lane * DD;
        float s = 0.f;
        for (int d = 0; d < DD; ++d) { float z = (float)zr[d]; s += z * z; }
        sqn[lane] = 0.5f * s;
    }
    __syncthreads();
    v16h a0 = load_a(A, DD, 0, lane);
    v16h a1 = load_a(A, DD, 32, lane);
    int hl = lane >> 4, c = lane & 15;
    size_t ob = (((size_t)b * HH + h) * NN + ntile * 16) * MM;
#pragma unroll
    for (int mt = 0; mt < 8; ++mt) {
        v8f acc = {};
        acc = wmma32(a0, load_b(Rt + (size_t)(mt * 16) * DD, DD, 0, lane), acc);
        acc = wmma32(a1, load_b(Rt + (size_t)(mt * 16) * DD, DD, 32, lane), acc);
#pragma unroll
        for (int i = 0; i < 8; ++i) {
            int r = i + (hl << 3);
            float v = __expf(acc[i] - sqn[r]) * MINV;
            qf[ob + (size_t)r * MM + mt * 16 + c] = (_Float16)v;
        }
    }
}

// ---------------------------------------------------------------------------
// kfT = (exp(z @ R - 0.5||z||^2) * M^-0.5)^T  stored [b,h,m,n] (K=n contiguous)
// out tile rows = m (A = R^T), cols = n (B = kh rows) -> per-lane col sqnorm.
// ---------------------------------------------------------------------------
__global__ void k_kft(const _Float16* __restrict__ kh, const _Float16* __restrict__ Rth,
                      _Float16* __restrict__ kfT) {
    int lane = threadIdx.x;
    int id = blockIdx.x;               // B*H*8*64 = 49152
    int ntile = id & 63; id >>= 6;
    int mtile = id & 7;  id >>= 3;
    int h = id % HH, b = id / HH;
    const _Float16* A  = Rth + ((size_t)h * MM + mtile * 16) * DD;
    const _Float16* Bt = kh + (((size_t)b * HH + h) * NN + ntile * 16) * DD;
    int hl = lane >> 4, c = lane & 15;
    const _Float16* zc = Bt + (size_t)c * DD;
    float s = 0.f;
    for (int d = 0; d < DD; ++d) { float z = (float)zc[d]; s += z * z; }
    s *= 0.5f;
    v8f acc = {};
    acc = wmma32(load_a(A, DD, 0, lane),  load_b(Bt, DD, 0, lane),  acc);
    acc = wmma32(load_a(A, DD, 32, lane), load_b(Bt, DD, 32, lane), acc);
    size_t base = ((size_t)b * HH + h) * MM * NN;
#pragma unroll
    for (int i = 0; i < 8; ++i) {
        int m = mtile * 16 + i + (hl << 3);
        int n = ntile * 16 + c;
        float v = __expf(acc[i] - s) * MINV;
        kfT[base + (size_t)m * NN + n] = (_Float16)v;
    }
}

// ksum[b,h,m] = sum_n kfT[b,h,m,n]
__global__ void k_ksum(const _Float16* __restrict__ kfT, float* __restrict__ ksum) {
    int i = blockIdx.x * 256 + threadIdx.x;   // B*H*M = 12288
    if (i >= BB * HH * MM) return;
    const _Float16* row = kfT + (size_t)i * NN;
    float s = 0.f;
    for (int n = 0; n < NN; ++n) s += (float)row[n];
    ksum[i] = s;
}

// KV = kfT @ v : [128,1024] x [64,1024]^T -> stored transposed KVt [b,h,d,m]
// Register double-buffered over the 32 k-steps.
__global__ void k_kv(const _Float16* __restrict__ kfT, const _Float16* __restrict__ vT,
                     _Float16* __restrict__ KVt) {
    int lane = threadIdx.x;
    int id = blockIdx.x;               // B*H*8*4 = 3072
    int dtile = id & 3; id >>= 2;
    int mtile = id & 7; id >>= 3;
    int bh = id;                        // 0..95
    const _Float16* A  = kfT + ((size_t)bh * MM + mtile * 16) * NN;
    const _Float16* Bt = vT  + ((size_t)bh * DD + dtile * 16) * NN;
    v16h a = load_a(A, NN, 0, lane);
    v16h b = load_b(Bt, NN, 0, lane);
    v8f acc = {};
    for (int k0 = 0; k0 < NN; k0 += 32) {
        int kn = (k0 + 32) & (NN - 1);
        v16h an = load_a(A, NN, kn, lane);
        v16h bn = load_b(Bt, NN, kn, lane);
        acc = wmma32(a, b, acc);
        a = an; b = bn;
    }
    int hl = lane >> 4, c = lane & 15;
#pragma unroll
    for (int i = 0; i < 8; ++i) {
        int m = mtile * 16 + i + (hl << 3);
        int d = dtile * 16 + c;
        KVt[((size_t)bh * DD + d) * MM + m] = (_Float16)acc[i];
    }
}

// rden[b,h,n] = 1 / max(qf[n,:] . ksum, 1e-12)   (L1 denom; attn is positive)
__global__ void k_den(const _Float16* __restrict__ qf, const float* __restrict__ ksum,
                      float* __restrict__ rden) {
    int i = blockIdx.x * 256 + threadIdx.x;   // B*H*N = 98304
    if (i >= BB * HH * NN) return;
    int bh = i >> 10;
    const _Float16* row = qf + (size_t)i * MM;
    const float* ks = ksum + (size_t)bh * MM;
    float s = 0.f;
    for (int m = 0; m < MM; ++m) s += (float)row[m] * ks[m];
    rden[i] = 1.0f / fmaxf(s, 1e-12f);
}

// out_attn = (qf @ KV) * rden, heads merged into K-contiguous [8192, 768] f16
__global__ void k_av(const _Float16* __restrict__ qf, const _Float16* __restrict__ KVt,
                     const float* __restrict__ rden, _Float16* __restrict__ aOut) {
    int lane = threadIdx.x;
    int id = blockIdx.x;               // B*H*(N/16) = 6144
    int ntile = id & 63; id >>= 6;
    int h = id % HH, b = id / HH;
    int bh = b * HH + h;
    const _Float16* A  = qf  + ((size_t)bh * NN + ntile * 16) * MM;
    const _Float16* Bt = KVt + (size_t)bh * DD * MM;
    __shared__ float rd[16];
    if (lane < 16) rd[lane] = rden[(size_t)bh * NN + ntile * 16 + lane];
    __syncthreads();
    v8f acc[4] = {};
    for (int k0 = 0; k0 < MM; k0 += 32) {
        v16h a = load_a(A, MM, k0, lane);
#pragma unroll
        for (int j = 0; j < 4; ++j)
            acc[j] = wmma32(a, load_b(Bt + (size_t)(j * 16) * MM, MM, k0, lane), acc[j]);
    }
    int hl = lane >> 4, c = lane & 15;
#pragma unroll
    for (int j = 0; j < 4; ++j)
#pragma unroll
        for (int i = 0; i < 8; ++i) {
            int r = i + (hl << 3);
            int n = ntile * 16 + r;
            int d = j * 16 + c;
            float v = acc[j][i] * rd[r];
            aOut[((size_t)b * NN + n) * CC + h * DD + d] = (_Float16)v;
        }
}

// final projection: [8192,768] x [768,768]^T + bias -> f32 out.
// Same 4-wave LDS-staged structure as k_qkv_gemm.
__global__ void k_proj(const _Float16* __restrict__ aOut, const _Float16* __restrict__ Ph,
                       const float* __restrict__ proj_b, float* __restrict__ out) {
    __shared__ __align__(16) _Float16 As[16 * CC];
    int lane = threadIdx.x & 31, wid = threadIdx.x >> 5;
    int row0 = blockIdx.x * 16;               // 512
    int col0 = blockIdx.y * 256 + wid * 64;   // 3 superblocks x 4 waves
    stage_tile(aOut + (size_t)row0 * CC, As, 16 * CC, threadIdx.x, 128);

    const _Float16* Bt[4];
#pragma unroll
    for (int j = 0; j < 4; ++j) Bt[j] = Ph + (size_t)(col0 + j * 16) * CC;

    v16h b[4], nb[4];
#pragma unroll
    for (int j = 0; j < 4; ++j) b[j] = load_b(Bt[j], CC, 0, lane);
    v8f acc[4] = {};
    for (int k0 = 0; k0 < CC; k0 += 32) {
        int kn = (k0 + 32 < CC) ? k0 + 32 : 0;
#pragma unroll
        for (int j = 0; j < 4; ++j) nb[j] = load_b(Bt[j], CC, kn, lane);
        v16h a = load_a(As, CC, k0, lane);
#pragma unroll
        for (int j = 0; j < 4; ++j) {
            acc[j] = wmma32(a, b[j], acc[j]);
            b[j] = nb[j];
        }
    }
    int hl = lane >> 4, c = lane & 15;
#pragma unroll
    for (int j = 0; j < 4; ++j) {
        int o = col0 + j * 16 + c;
        float bias = proj_b[o];
#pragma unroll
        for (int i = 0; i < 8; ++i) {
            int r = row0 + i + (hl << 3);
            out[(size_t)r * CC + o] = acc[j][i] + bias;
        }
    }
}

// ---------------------------------------------------------------------------
extern "C" void kernel_launch(void* const* d_in, const int* in_sizes, int n_in,
                              void* d_out, int out_size, void* d_ws, size_t ws_size,
                              hipStream_t stream) {
    const float* x      = (const float*)d_in[0];  // [8,1024,768]
    const float* qkv_w  = (const float*)d_in[1];  // [2304,768]
    const float* proj_w = (const float*)d_in[2];  // [768,768]
    const float* proj_b = (const float*)d_in[3];  // [768]
    const float* rand_m = (const float*)d_in[4];  // [12,64,128]
    float* out = (float*)d_out;

    char* ws = (char*)d_ws;
    size_t off = 0;
    auto alloc = [&](size_t bytes) -> void* {
        void* p = ws + off;
        off = (off + bytes + 255) & ~(size_t)255;
        return p;
    };
    const size_t ROWS = (size_t)BB * NN;  // 8192
    _Float16* Xh   = (_Float16*)alloc(ROWS * CC * 2);
    _Float16* Wh   = (_Float16*)alloc((size_t)3 * CC * CC * 2);
    _Float16* Ph   = (_Float16*)alloc((size_t)CC * CC * 2);
    _Float16* Rth  = (_Float16*)alloc((size_t)HH * MM * DD * 2);
    _Float16* qh   = (_Float16*)alloc((size_t)BB * HH * NN * DD * 2);
    _Float16* kh   = (_Float16*)alloc((size_t)BB * HH * NN * DD * 2);
    _Float16* vT   = (_Float16*)alloc((size_t)BB * HH * DD * NN * 2);
    _Float16* qf   = (_Float16*)alloc((size_t)BB * HH * NN * MM * 2);
    _Float16* kfT  = (_Float16*)alloc((size_t)BB * HH * MM * NN * 2);
    float*    ksum = (float*)   alloc((size_t)BB * HH * MM * 4);
    _Float16* KVt  = (_Float16*)alloc((size_t)BB * HH * DD * MM * 2);
    float*    rden = (float*)   alloc((size_t)BB * HH * NN * 4);
    _Float16* aOut = (_Float16*)alloc(ROWS * CC * 2);

    int nx = (int)(ROWS * CC);
    k_cvt<<<(nx + 255) / 256, 256, 0, stream>>>(x, Xh, nx);
    int nw = 3 * CC * CC;
    k_cvt<<<(nw + 255) / 256, 256, 0, stream>>>(qkv_w, Wh, nw);
    int np = CC * CC;
    k_cvt<<<(np + 255) / 256, 256, 0, stream>>>(proj_w, Ph, np);
    k_cvt_randT<<<(HH * DD * MM + 255) / 256, 256, 0, stream>>>(rand_m, Rth);

    k_qkv_gemm<<<dim3(ROWS / 16, (3 * CC) / 256), 128, 0, stream>>>(Xh, Wh, qh, kh, vT);
    k_qf <<<BB * HH * (NN / 16), 32, 0, stream>>>(qh, Rth, qf);
    k_kft<<<BB * HH * (MM / 16) * (NN / 16), 32, 0, stream>>>(kh, Rth, kfT);
    k_ksum<<<(BB * HH * MM + 255) / 256, 256, 0, stream>>>(kfT, ksum);
    k_kv <<<BB * HH * (MM / 16) * (DD / 16), 32, 0, stream>>>(kfT, vT, KVt);
    k_den<<<(BB * HH * NN + 255) / 256, 256, 0, stream>>>(qf, ksum, rden);
    k_av <<<BB * HH * (NN / 16), 32, 0, stream>>>(qf, KVt, rden, aOut);
    k_proj<<<dim3(ROWS / 16, CC / 256), 128, 0, stream>>>(aOut, Ph, proj_b, out);
}
